// RNN_45543833207230
// MI455X (gfx1250) — matrix operand order repaired and checked by reading, not compiled, not run on hardware
//
#include <hip/hip_runtime.h>
#include <hip/hip_bf16.h>

// ---------------------------------------------------------------------------
// Stacked tanh-RNN for MI455X (gfx1250, wave32, WMMA).
//  - All GEMMs on v_wmma_f32_16x16x32_bf16 (fp32 accumulate).
//  - Recurrent sweep: per-step kernels (graph-replayed), 1 wave = 1 16x16 tile,
//    K-loop split across the [h | x] concat, dual accumulators to break the
//    WMMA dependency chain.
//  - Logits: batched [8192,1024]x[1024,10000] WMMA GEMM, LDS tiles with
//    register-staged global->LDS pipelining.
// ---------------------------------------------------------------------------

typedef __attribute__((ext_vector_type(16))) __bf16 v16bf;
typedef __attribute__((ext_vector_type(8)))  __bf16 v8bf;
typedef __attribute__((ext_vector_type(8)))  float  v8f;

#define EMB   1024
#define HID   1024
#define SEQ   128
#define BATCH 64
#define VOCAB 10000
#define TB    (SEQ * BATCH)   // 8192

// ---------------------------------------------------------------------------
// Fragment builders (ISA 7.12.2 layouts).
// A (16x32 bf16, MxK): lane<16 -> row=lane,   K = {0..7, 16..23}
//                      lane>=16 -> row=lane-16, K = {8..15, 24..31}
//   -> per-lane base = row*lda + (lane>=16 ? 8 : 0); hi part at +16.
// B (32x16 bf16, KxN): lane<16 -> col=lane, K = {0..7, 16..23}; lane>=16 -> K = {8..15, 24..31}
//   -> per-lane base = col*ldb + (lane>=16 ? 8 : 0); hi part at +16.
// C/D (16x16 f32): 8 VGPRs, N=lane%16, M = vgpr + (lane>=16 ? 8 : 0)
// ---------------------------------------------------------------------------
__device__ __forceinline__ v16bf frag16(const __bf16* __restrict__ p, int hiOff) {
    v8bf lo = *(const v8bf*)p;
    v8bf hi = *(const v8bf*)(p + hiOff);
    return __builtin_shufflevector(lo, hi, 0, 1, 2, 3, 4, 5, 6, 7,
                                           8, 9, 10, 11, 12, 13, 14, 15);
}

__device__ __forceinline__ v8f wmma_bf16(v16bf a, v16bf b, v8f c) {
    return __builtin_amdgcn_wmma_f32_16x16x32_bf16(
        /*neg_a=*/false, a, /*neg_b=*/false, b,
        /*c_mod=*/(short)0, c, /*reuse_a=*/false, /*reuse_b=*/false);
}

// ---------------------------------------------------------------------------
// fp32 -> bf16 conversion (weights, initial hidden)
// ---------------------------------------------------------------------------
__global__ void cvt_f32_to_bf16(const float* __restrict__ src,
                                __bf16* __restrict__ dst, int n) {
    int i = blockIdx.x * blockDim.x + threadIdx.x;
    int stride = gridDim.x * blockDim.x;
    for (; i < n; i += stride) dst[i] = (__bf16)src[i];
}

// ---------------------------------------------------------------------------
// Embedding gather: [T,B] tokens -> bf16 [T*B, EMB]
// ---------------------------------------------------------------------------
__global__ void embed_gather(const int* __restrict__ tok,
                             const float* __restrict__ table,
                             __bf16* __restrict__ xemb) {
    int tb = blockIdx.x;                 // 0 .. T*B-1 (t-major, matches [T,B])
    long idx = (long)tok[tb];
    const float* row = table + idx * EMB;
    __bf16* out = xemb + (long)tb * EMB;
    for (int e = threadIdx.x; e < EMB; e += blockDim.x)
        out[e] = (__bf16)row[e];
}

// ---------------------------------------------------------------------------
// One recurrent layer step: out = tanh([Hprev | Xin] @ W^T + bias)
// M=64 (batch), N=1024 (hidden), K=2048. 256 16x16 tiles, 1 per wave,
// 8 waves/block, 32 blocks. Weights are L2-resident bf16.
// K-loop split at the concat boundary; dual accumulators hide WMMA hazards.
// ---------------------------------------------------------------------------
__global__ __launch_bounds__(256)
void rnn_layer(const __bf16* __restrict__ Hprev,   // [64,1024] bf16
               const __bf16* __restrict__ Xin,     // [64,1024] bf16
               const __bf16* __restrict__ W,       // [1024,2048] bf16 row-major
               const float*  __restrict__ bias,    // [1024]
               __bf16* __restrict__ Hnext,         // [64,1024] bf16 (next step)
               float*  __restrict__ Hf32,          // [64,1024] f32 (final slot)
               __bf16* __restrict__ Yextra) {      // optional [64,1024] bf16
    const int lane = threadIdx.x & 31;
    const int wave = threadIdx.x >> 5;
    const int tile = blockIdx.x * 8 + wave;  // 0..255
    const int m0 = (tile & 3) * 16;          // batch-row tile
    const int n0 = (tile >> 2) * 16;         // hidden-col tile

    // Hoisted per-lane fragment base offsets (constant strides -> imm offsets)
    const int aoff = (lane & 15) * HID + ((lane >> 4) << 3);
    const int boff = (lane & 15) * (2 * HID) + ((lane >> 4) << 3);
    const __bf16* Ah = Hprev + m0 * HID + aoff;
    const __bf16* Ax = Xin   + m0 * HID + aoff;
    const __bf16* Wh = W + (long)n0 * (2 * HID) + boff;        // K 0..1023
    const __bf16* Wx = Wh + HID;                                // K 1024..2047

    v8f acc0 = {}, acc1 = {};
    #pragma unroll 4
    for (int k0 = 0; k0 < HID; k0 += 64) {
        v16bf a0 = frag16(Ah + k0,      16);
        v16bf b0 = frag16(Wh + k0,      16);
        v16bf a1 = frag16(Ah + k0 + 32, 16);
        v16bf b1 = frag16(Wh + k0 + 32, 16);
        acc0 = wmma_bf16(a0, b0, acc0);
        acc1 = wmma_bf16(a1, b1, acc1);
    }
    #pragma unroll 4
    for (int k0 = 0; k0 < HID; k0 += 64) {
        v16bf a0 = frag16(Ax + k0,      16);
        v16bf b0 = frag16(Wx + k0,      16);
        v16bf a1 = frag16(Ax + k0 + 32, 16);
        v16bf b1 = frag16(Wx + k0 + 32, 16);
        acc0 = wmma_bf16(a0, b0, acc0);
        acc1 = wmma_bf16(a1, b1, acc1);
    }
    v8f acc = acc0 + acc1;

    const int n  = n0 + (lane & 15);
    const int mb = m0 + ((lane >> 4) << 3);
    const float bn = bias[n];
    #pragma unroll
    for (int v = 0; v < 8; ++v) {
        float val = tanhf(acc[v] + bn);
        int m = mb + v;
        Hf32[m * HID + n]  = val;
        __bf16 bv = (__bf16)val;
        Hnext[m * HID + n] = bv;
        if (Yextra) Yextra[m * HID + n] = bv;
    }
}

// ---------------------------------------------------------------------------
// Batched logits GEMM: [8192,1024](bf16) x Wy^T [1024,10000](bf16) + by -> f32
// Block tile 64(M) x 128(N); 8 waves, each owns 1 M-subtile x 4 N-subtiles.
// K staged through LDS in 32-wide slabs; next slab's global loads are issued
// into registers while WMMAs consume the current LDS tile.
// ---------------------------------------------------------------------------
#define LT_M 64
#define LT_N 128
#define LDS_STRIDE 40  // 32 bf16 + 8 pad; 80B rows keep 16B alignment

__global__ __launch_bounds__(256)
void logits_gemm(const __bf16* __restrict__ Ya,   // [8192,1024] bf16
                 const __bf16* __restrict__ Wyb,  // [10000,1024] bf16
                 const float*  __restrict__ by,   // [10000]
                 float* __restrict__ out) {       // [8192,10000] f32
    __shared__ __bf16 As[LT_M * LDS_STRIDE];      // 5.0 KB
    __shared__ __bf16 Bs[LT_N * LDS_STRIDE];      // 10.0 KB

    const int tid   = threadIdx.x;
    const int lane  = tid & 31;
    const int wave  = tid >> 5;
    const int mBase = blockIdx.x * LT_M;
    const int nBase = blockIdx.y * LT_N;
    const int mSub  = wave & 3;    // 16-row strip within block tile
    const int nGrp  = wave >> 2;   // 0/1 -> four consecutive 16-col subtiles

    unsigned int* AsU = (unsigned int*)As;
    unsigned int* BsU = (unsigned int*)Bs;
    const unsigned int* Yu = (const unsigned int*)Ya;
    const unsigned int* Wu = (const unsigned int*)Wyb;

    // Per-thread staging slots: A tile 64x16 dwords (4/thread),
    //                           B tile 128x16 dwords (8/thread)
    const int ar = tid >> 4, ac = tid & 15;   // +r stride 16 per chunk
    unsigned int aReg[4], bReg[8];

    auto stageLoad = [&](int k0) {
        #pragma unroll
        for (int i = 0; i < 4; ++i) {
            int r = ar + i * 16;
            aReg[i] = Yu[((long)(mBase + r) * HID + k0) / 2 + ac];
        }
        #pragma unroll
        for (int i = 0; i < 8; ++i) {
            int r = ar + i * 16;
            int n = nBase + r;
            bReg[i] = (n < VOCAB) ? Wu[((long)n * HID + k0) / 2 + ac] : 0u;
        }
    };

    v8f acc[4] = {};
    stageLoad(0);  // prologue

    for (int k0 = 0; k0 < HID; k0 += 32) {
        // Commit staged registers to LDS
        #pragma unroll
        for (int i = 0; i < 4; ++i)
            AsU[(ar + i * 16) * (LDS_STRIDE / 2) + ac] = aReg[i];
        #pragma unroll
        for (int i = 0; i < 8; ++i)
            BsU[(ar + i * 16) * (LDS_STRIDE / 2) + ac] = bReg[i];
        __syncthreads();

        // Issue next slab's global loads before compute (latency overlap)
        if (k0 + 32 < HID) stageLoad(k0 + 32);

        v16bf a = frag16(As + (mSub * 16 + (lane & 15)) * LDS_STRIDE
                            + ((lane >> 4) << 3), 16);
        #pragma unroll
        for (int i = 0; i < 4; ++i) {
            v16bf b = frag16(Bs + (nGrp * 64 + i * 16 + (lane & 15)) * LDS_STRIDE
                                + ((lane >> 4) << 3), 16);
            acc[i] = wmma_bf16(a, b, acc[i]);
        }
        __syncthreads();
    }

    const int mRow = mBase + mSub * 16 + ((lane >> 4) << 3);
    #pragma unroll
    for (int i = 0; i < 4; ++i) {
        int n = nBase + nGrp * 64 + i * 16 + (lane & 15);
        if (n < VOCAB) {
            float bn = by[n];
            #pragma unroll
            for (int v = 0; v < 8; ++v)
                out[(long)(mRow + v) * VOCAB + n] = acc[i][v] + bn;
        }
    }
}

// ---------------------------------------------------------------------------
// Host orchestration
// ---------------------------------------------------------------------------
extern "C" void kernel_launch(void* const* d_in, const int* in_sizes, int n_in,
                              void* d_out, int out_size, void* d_ws, size_t ws_size,
                              hipStream_t stream) {
    (void)in_sizes; (void)n_in; (void)out_size; (void)ws_size;

    const int*   tok  = (const int*)d_in[0];
    const float* hid0 = (const float*)d_in[1];
    const float* etab = (const float*)d_in[2];
    const float* W0   = (const float*)d_in[3];
    const float* b0   = (const float*)d_in[4];
    const float* Wrec = (const float*)d_in[5];   // [1,1024,2048]
    const float* brec = (const float*)d_in[6];   // [1,1024]
    const float* Wy   = (const float*)d_in[7];
    const float* by   = (const float*)d_in[8];

    float* logits = (float*)d_out;                       // [128,64,10000]
    float* hfin   = logits + (long)TB * VOCAB;           // [2,64,1024]

    // Workspace carve (256B aligned): ~63 MB total
    char*  ws  = (char*)d_ws;
    size_t off = 0;
    auto carve = [&](size_t bytes) -> char* {
        char* p = ws + off;
        off = (off + bytes + 255) & ~(size_t)255;
        return p;
    };
    __bf16* W0b  = (__bf16*)carve((size_t)HID * 2 * HID * 2);       // 4 MB
    __bf16* Wrb  = (__bf16*)carve((size_t)HID * 2 * HID * 2);       // 4 MB
    __bf16* Wyb  = (__bf16*)carve((size_t)VOCAB * HID * 2);         // 20 MB
    __bf16* Xemb = (__bf16*)carve((size_t)TB * EMB * 2);            // 16 MB
    __bf16* Ybuf = (__bf16*)carve((size_t)TB * HID * 2);            // 16 MB
    __bf16* hb[2][2];
    for (int l = 0; l < 2; ++l)
        for (int p = 0; p < 2; ++p)
            hb[l][p] = (__bf16*)carve((size_t)BATCH * HID * 2);     // 4x128KB

    // 1) Weight + initial-hidden conversion to bf16 (L2-resident afterwards)
    cvt_f32_to_bf16<<<1024, 256, 0, stream>>>(W0,   W0b, HID * 2 * HID);
    cvt_f32_to_bf16<<<1024, 256, 0, stream>>>(Wrec, Wrb, HID * 2 * HID);
    cvt_f32_to_bf16<<<2048, 256, 0, stream>>>(Wy,   Wyb, VOCAB * HID);
    cvt_f32_to_bf16<<<128,  256, 0, stream>>>(hid0,               hb[0][0], BATCH * HID);
    cvt_f32_to_bf16<<<128,  256, 0, stream>>>(hid0 + BATCH * HID, hb[1][0], BATCH * HID);

    // 2) Embedding gather for all timesteps
    embed_gather<<<TB, 256, 0, stream>>>(tok, etab, Xemb);

    // 3) Sequential recurrent sweep (kernel launch = global barrier per step);
    //    hidden double-buffered across steps to avoid cross-block RAW.
    for (int t = 0; t < SEQ; ++t) {
        int pr = t & 1, nx = pr ^ 1;
        rnn_layer<<<32, 256, 0, stream>>>(
            hb[0][pr], Xemb + (size_t)t * BATCH * EMB, W0b, b0,
            hb[0][nx], hfin + 0 * BATCH * HID, (__bf16*)nullptr);
        rnn_layer<<<32, 256, 0, stream>>>(
            hb[1][pr], hb[0][nx], Wrb, brec,
            hb[1][nx], hfin + 1 * BATCH * HID,
            Ybuf + (size_t)t * BATCH * HID);
    }

    // 4) Batched logits projection: dominant 168 GFLOP GEMM on WMMA
    dim3 lgrid(TB / LT_M, (VOCAB + LT_N - 1) / LT_N);   // 128 x 79
    logits_gemm<<<lgrid, 256, 0, stream>>>(Ybuf, Wyb, by, logits);
}
